// Net_1554778161656
// MI455X (gfx1250) — matrix operand order repaired and checked by reading, not compiled, not run on hardware
//
#include <hip/hip_runtime.h>
#include <hip/hip_bf16.h>
#include <math.h>

typedef __attribute__((ext_vector_type(16))) __bf16 v16bf;
typedef __attribute__((ext_vector_type(8)))  __bf16 v8bf;
typedef __attribute__((ext_vector_type(8)))  float  v8f;

// ---------------- workspace layout (bytes) ----------------
// 0       : uint state[4][2]   (radix-select {prefix, remaining} per score tensor)
// 256     : uint hist[256]
// 2048    : w1m   f32 [32*9]              (masked conv1 weights)
// 4096    : fc2m  f32 [10*128]            (masked fc2 weights)
// 16384   : w2pk  bf16[9*4*32*16]         (conv2 weights, WMMA B-fragment layout)
// 65536   : fc1pk bf16[8*288*32*16]       (fc1 weights, WMMA B-fragment layout, NHWC-permuted K)
// 4  MB   : h1    bf16 NHWC [2048][26][26][32]   (88.6 MB, L2-resident)
// 96 MB   : pool  bf16 NHWC [2048*12*12][64]     (37.7 MB, L2-resident)
// 136 MB  : hfc1  f32  [4 kchunks][2048][128]    (4 MB, deterministic K-split partials)
// total needed ~140 MB of d_ws.

__device__ __forceinline__ unsigned short f2bf(float f) {
  unsigned u = __float_as_uint(f);
  u += 0x7FFFu + ((u >> 16) & 1u);   // round-to-nearest-even
  return (unsigned short)(u >> 16);
}

// ---------------- radix select (k-th smallest of |s|) ----------------
__global__ void init_state_kernel(unsigned* state) {
  // remaining = j = n/2 (number of lowest-|score| entries zeroed)
  state[0] = 0; state[1] = 144;      // conv1_s  n=288
  state[2] = 0; state[3] = 9216;     // conv2_s  n=18432
  state[4] = 0; state[5] = 589824;   // fc1_s    n=1179648
  state[6] = 0; state[7] = 640;      // fc2_s    n=1280
}

__global__ void zero_hist_kernel(unsigned* hist) { hist[threadIdx.x] = 0u; }

__global__ void hist_kernel(const float* __restrict__ s, int n,
                            const unsigned* __restrict__ state,
                            unsigned* __restrict__ hist, int pass) {
  unsigned prefix = state[0];
  int stride = gridDim.x * blockDim.x;
  for (int i = blockIdx.x * blockDim.x + threadIdx.x; i < n; i += stride) {
    unsigned key = __float_as_uint(fabsf(s[i]));
    bool match = (pass == 3) || (((key ^ prefix) >> (8 * (pass + 1))) == 0u);
    if (match) atomicAdd(&hist[(key >> (8 * pass)) & 0xFFu], 1u);
  }
}

__global__ void scan_kernel(unsigned* state, const unsigned* __restrict__ hist, int pass) {
  unsigned rem = state[1], c = 0;
  for (int bin = 0; bin < 256; ++bin) {
    unsigned h = hist[bin];
    if (c + h > rem) {
      state[0] |= ((unsigned)bin) << (8 * pass);
      state[1] = rem - c;
      return;
    }
    c += h;
  }
}

// ---------------- masked-weight packing ----------------
__global__ void apply_mask_kernel(const float* __restrict__ w, const float* __restrict__ s,
                                  const unsigned* __restrict__ state, float* __restrict__ out, int n) {
  int i = blockIdx.x * blockDim.x + threadIdx.x;
  if (i >= n) return;
  unsigned thr = state[0];
  unsigned key = __float_as_uint(fabsf(s[i]));
  out[i] = (key >= thr) ? w[i] : 0.0f;
}

// conv2 weights [64][32][3][3] -> B fragments: [(kykx*4+g)*32+lane][e] ; k=ic, n=oc
__global__ void pack_w2_kernel(const float* __restrict__ w, const float* __restrict__ s,
                               const unsigned* __restrict__ state, unsigned short* __restrict__ out) {
  int idx = blockIdx.x * blockDim.x + threadIdx.x;   // < 18432
  unsigned thr = state[0];
  int e = idx & 15, lane = (idx >> 4) & 31, tmp = idx >> 9;
  int g = tmp & 3, kykx = tmp >> 2;
  int oc = g * 16 + (lane & 15);
  int ic = ((lane >> 4) << 3) + (e & 7) + ((e >> 3) << 4);
  int wi = (oc * 32 + ic) * 9 + kykx;
  unsigned key = __float_as_uint(fabsf(s[wi]));
  float v = (key >= thr) ? w[wi] : 0.0f;
  out[idx] = f2bf(v);
}

// fc1 weights [128][9216 (NCHW flat)] -> B fragments with NHWC K permutation
__global__ void pack_fc1_kernel(const float* __restrict__ w, const float* __restrict__ s,
                                const unsigned* __restrict__ state, unsigned short* __restrict__ out) {
  int idx = blockIdx.x * blockDim.x + threadIdx.x;   // < 1179648
  unsigned thr = state[0];
  int e = idx & 15, lane = (idx >> 4) & 31, tmp = idx >> 9;
  int t = tmp % 288, g = tmp / 288;
  int k = t * 32 + ((lane >> 4) << 3) + (e & 7) + ((e >> 3) << 4);           // NHWC k' index
  int n = g * 16 + (lane & 15);
  int ch = k & 63, pos = k >> 6;                     // pos = py*12+px
  int c = ch * 144 + pos;                            // reference NCHW flatten
  int wi = n * 9216 + c;
  unsigned key = __float_as_uint(fabsf(s[wi]));
  float v = (key >= thr) ? w[wi] : 0.0f;
  out[idx] = f2bf(v);
}

// ---------------- conv1 (fp32, direct) + ReLU -> NHWC bf16 ----------------
__global__ __launch_bounds__(256) void conv1_kernel(const float* __restrict__ x,
                                                    const float* __restrict__ w1m,
                                                    unsigned short* __restrict__ h1) {
  __shared__ float sw[288];
  for (int i = threadIdx.x; i < 288; i += 256) sw[i] = w1m[i];
  __syncthreads();
  int tid = blockIdx.x * 256 + threadIdx.x;          // < 44,302,336
  int oc = tid & 31, rest = tid >> 5;                // rest = b*676 + yo*26 + xo
  int xo = rest % 26, yo = (rest / 26) % 26, b = rest / 676;
  const float* xb = x + b * 784;
  float acc = 0.f;
#pragma unroll
  for (int ky = 0; ky < 3; ++ky)
#pragma unroll
    for (int kx = 0; kx < 3; ++kx)
      acc += xb[(yo + ky) * 28 + xo + kx] * sw[oc * 9 + ky * 3 + kx];
  h1[(size_t)rest * 32 + oc] = f2bf(fmaxf(acc, 0.f));
}

// ---------------- conv2 (implicit GEMM, bf16 WMMA) + fused 2x2 maxpool ----------------
__global__ __launch_bounds__(256) void conv2_pool_kernel(const unsigned short* __restrict__ h1,
                                                         const unsigned short* __restrict__ w2pk,
                                                         unsigned short* __restrict__ pool) {
  __shared__ __align__(16) unsigned short ldsb[18432];   // 36,864 B of packed B fragments
  {
    // Async global->LDS staging (CDNA5 GLOBAL_LOAD_ASYNC_TO_LDS_B128, ASYNCcnt-tracked).
    // Low 32 bits of a flat shared-aperture address are the LDS byte offset (ISA 10.2).
    unsigned ldsbase = (unsigned)(unsigned long long)(void*)&ldsb[0];
    unsigned long long gbase = (unsigned long long)(const void*)w2pk;
    for (int i = threadIdx.x; i < 2304; i += 256) {      // 2304 = 9 full passes of 256 lanes
      unsigned laddr = ldsbase + (unsigned)i * 16u;
      unsigned long long gaddr = gbase + (unsigned long long)i * 16ull;
      asm volatile("global_load_async_to_lds_b128 %0, %1, off"
                   :: "v"(laddr), "v"(gaddr) : "memory");
    }
    asm volatile("s_wait_asynccnt 0x0" ::: "memory");
  }
  __syncthreads();

  const int wid = threadIdx.x >> 5;
  const int lane = threadIdx.x & 31;
  const int tile = blockIdx.x * 8 + wid;               // < 73728 M-tiles of 16
  const int m = tile * 16 + (lane & 15);               // m = (b,py,px,quad)
  const int q = m & 3, pidx = m >> 2;
  const int px = pidx % 12, py = (pidx / 12) % 12, b = pidx / 144;
  const int oy = 2 * py + (q >> 1), ox = 2 * px + (q & 1);
  const int halfsel = (lane >> 4) << 3;                // 0 or 8 bf16 elements

  v8f z = {0.f, 0.f, 0.f, 0.f, 0.f, 0.f, 0.f, 0.f};
  v8f acc[4] = {z, z, z, z};
  const v16bf* bbase = (const v16bf*)ldsb;

#pragma unroll
  for (int kk = 0; kk < 9; ++kk) {
    const int ky = kk / 3, kx = kk % 3;
    const unsigned short* row = h1 + (size_t)(((b * 26) + (oy + ky)) * 26 + (ox + kx)) * 32;
    v8bf alo = *(const v8bf*)(row + halfsel);
    v8bf ahi = *(const v8bf*)(row + halfsel + 16);
    v16bf a = __builtin_shufflevector(alo, ahi, 0, 1, 2, 3, 4, 5, 6, 7,
                                      8, 9, 10, 11, 12, 13, 14, 15);
#pragma unroll
    for (int g = 0; g < 4; ++g) {
      v16bf bf = bbase[(kk * 4 + g) * 32 + lane];
      acc[g] = __builtin_amdgcn_wmma_f32_16x16x32_bf16(false, a, false, bf,
                                                       (short)0, acc[g], false, false);
    }
  }

  // 2x2 maxpool in the D epilogue: consecutive M quads live in acc elements.
  const int pmb = tile * 4 + ((lane >> 4) << 1);
#pragma unroll
  for (int g = 0; g < 4; ++g) {
    float m0 = fmaxf(fmaxf(acc[g][0], acc[g][1]), fmaxf(acc[g][2], acc[g][3]));
    float m1 = fmaxf(fmaxf(acc[g][4], acc[g][5]), fmaxf(acc[g][6], acc[g][7]));
    int n = g * 16 + (lane & 15);
    pool[(size_t)(pmb + 0) * 64 + n] = f2bf(m0);
    pool[(size_t)(pmb + 1) * 64 + n] = f2bf(m1);
  }
}

// ---------------- fc1 (bf16 WMMA GEMM 2048x128x9216, K split 4-way) ----------------
// Each wave: M16 x N32 tile (two N-groups share one A fragment), 72 k-steps.
// Partials written to per-chunk buffers (deterministic; no fp atomics).
__global__ __launch_bounds__(256) void fc1_kernel(const unsigned short* __restrict__ pool,
                                                  const unsigned short* __restrict__ fc1pk,
                                                  float* __restrict__ hfc1) {
  const int wid = threadIdx.x >> 5;
  const int lane = threadIdx.x & 31;
  const int tile = blockIdx.x * 8 + wid;               // < 2048
  const int kc = tile & 3;                             // K chunk (72 steps)
  const int gp = (tile >> 2) & 3;                      // N-group pair -> groups 2gp, 2gp+1
  const int mt = tile >> 4;                            // 0..127
  const int m = mt * 16 + (lane & 15);
  const unsigned short* arow = pool + (size_t)m * 9216;
  const int halfsel = (lane >> 4) << 3;
  const v16bf* bbase = (const v16bf*)fc1pk;

  v8f z = {0.f, 0.f, 0.f, 0.f, 0.f, 0.f, 0.f, 0.f};
  v8f acc0 = z, acc1 = z;
  const int t0 = kc * 72;
  for (int t = t0; t < t0 + 72; ++t) {
    if ((t & 7) == 0)
      __builtin_prefetch(arow + (t + 8) * 32, 0, 1);   // global_prefetch_b8
    v8bf alo = *(const v8bf*)(arow + t * 32 + halfsel);
    v8bf ahi = *(const v8bf*)(arow + t * 32 + halfsel + 16);
    v16bf a = __builtin_shufflevector(alo, ahi, 0, 1, 2, 3, 4, 5, 6, 7,
                                      8, 9, 10, 11, 12, 13, 14, 15);
    v16bf b0 = bbase[((2 * gp + 0) * 288 + t) * 32 + lane];
    v16bf b1 = bbase[((2 * gp + 1) * 288 + t) * 32 + lane];
    acc0 = __builtin_amdgcn_wmma_f32_16x16x32_bf16(false, a, false, b0,
                                                   (short)0, acc0, false, false);
    acc1 = __builtin_amdgcn_wmma_f32_16x16x32_bf16(false, a, false, b1,
                                                   (short)0, acc1, false, false);
  }

  float* part = hfc1 + (size_t)kc * 262144;            // [2048][128] partial for this chunk
  const int col0 = (2 * gp) * 16 + (lane & 15);
  const int rbase = mt * 16 + ((lane >> 4) << 3);
#pragma unroll
  for (int r = 0; r < 8; ++r) {
    part[(size_t)(rbase + r) * 128 + col0]      = acc0[r];
    part[(size_t)(rbase + r) * 128 + col0 + 16] = acc1[r];
  }
}

// ---------------- fc2 + log_softmax (one wave32 per row; sums K-split partials + ReLU) ----------------
__global__ __launch_bounds__(256) void fc2_softmax_kernel(const float* __restrict__ hfc1,
                                                          const float* __restrict__ fc2m,
                                                          float* __restrict__ out) {
  const int wid = threadIdx.x >> 5;
  const int lane = threadIdx.x & 31;
  const int row = blockIdx.x * 8 + wid;                // < 2048
  float hv[4];
#pragma unroll
  for (int i = 0; i < 4; ++i) {
    size_t off = (size_t)row * 128 + i * 32 + lane;
    float v = hfc1[off] + hfc1[262144 + off] + hfc1[2 * 262144 + off] + hfc1[3 * 262144 + off];
    hv[i] = fmaxf(v, 0.f);                             // ReLU after full K sum
  }

  float logit[10];
#pragma unroll
  for (int n = 0; n < 10; ++n) {
    float p = 0.f;
#pragma unroll
    for (int i = 0; i < 4; ++i) p += hv[i] * fc2m[n * 128 + i * 32 + lane];
#pragma unroll
    for (int off = 16; off > 0; off >>= 1) p += __shfl_xor(p, off, 32);
    logit[n] = p;
  }
  float mx = logit[0];
#pragma unroll
  for (int n = 1; n < 10; ++n) mx = fmaxf(mx, logit[n]);
  float sum = 0.f;
#pragma unroll
  for (int n = 0; n < 10; ++n) sum += expf(logit[n] - mx);
  float lse = logf(sum);
  if (lane == 0) {
#pragma unroll
    for (int n = 0; n < 10; ++n) out[(size_t)row * 10 + n] = logit[n] - mx - lse;
  }
}

// ---------------- driver ----------------
extern "C" void kernel_launch(void* const* d_in, const int* in_sizes, int n_in,
                              void* d_out, int out_size, void* d_ws, size_t ws_size,
                              hipStream_t stream) {
  const float* x   = (const float*)d_in[0];
  const float* c1w = (const float*)d_in[1];
  const float* c1s = (const float*)d_in[2];
  const float* c2w = (const float*)d_in[3];
  const float* c2s = (const float*)d_in[4];
  const float* f1w = (const float*)d_in[5];
  const float* f1s = (const float*)d_in[6];
  const float* f2w = (const float*)d_in[7];
  const float* f2s = (const float*)d_in[8];

  char* ws = (char*)d_ws;
  unsigned*       state = (unsigned*)(ws + 0);
  unsigned*       hist  = (unsigned*)(ws + 256);
  float*          w1m   = (float*)(ws + 2048);
  float*          fc2m  = (float*)(ws + 4096);
  unsigned short* w2pk  = (unsigned short*)(ws + 16384);
  unsigned short* fc1pk = (unsigned short*)(ws + 65536);
  unsigned short* h1    = (unsigned short*)(ws + (4ull << 20));
  unsigned short* pool  = (unsigned short*)(ws + (96ull << 20));
  float*          hfc1  = (float*)(ws + (136ull << 20));

  init_state_kernel<<<1, 1, 0, stream>>>(state);

  const float* sc[4] = {c1s, c2s, f1s, f2s};
  const int    sn[4] = {288, 18432, 1179648, 1280};
  for (int t = 0; t < 4; ++t) {
    for (int pass = 3; pass >= 0; --pass) {
      zero_hist_kernel<<<1, 256, 0, stream>>>(hist);
      int blocks = (sn[t] + 255) / 256;
      if (blocks > 2048) blocks = 2048;
      hist_kernel<<<blocks, 256, 0, stream>>>(sc[t], sn[t], state + 2 * t, hist, pass);
      scan_kernel<<<1, 1, 0, stream>>>(state + 2 * t, hist, pass);
    }
  }

  apply_mask_kernel<<<2, 256, 0, stream>>>(c1w, c1s, state + 0, w1m, 288);
  apply_mask_kernel<<<5, 256, 0, stream>>>(f2w, f2s, state + 6, fc2m, 1280);
  pack_w2_kernel<<<72, 256, 0, stream>>>(c2w, c2s, state + 2, w2pk);
  pack_fc1_kernel<<<4608, 256, 0, stream>>>(f1w, f1s, state + 4, fc1pk);

  conv1_kernel<<<173056, 256, 0, stream>>>(x, w1m, h1);
  conv2_pool_kernel<<<9216, 256, 0, stream>>>(h1, w2pk, pool);
  fc1_kernel<<<256, 256, 0, stream>>>(pool, fc1pk, hfc1);
  fc2_softmax_kernel<<<256, 256, 0, stream>>>(hfc1, fc2m, (float*)d_out);
}
